// MoELayer_10651518894848
// MI455X (gfx1250) — matrix operand order repaired and checked by reading, not compiled, not run on hardware
//
#include <hip/hip_runtime.h>
#include <hip/hip_bf16.h>

typedef __attribute__((ext_vector_type(16))) __bf16 v16bf;
typedef __attribute__((ext_vector_type(8)))  float  v8f;

#define B_TOK 8192
#define D_IN  1024
#define H_DIM 4096
#define O_DIM 1024
#define NEXP  10   // 2 shared + 8 routed, gate-column order

// fp32 pair -> packed bf16 dword in ONE instruction. The gfx1250 assembler
// accepts v_cvt_pk_bf16_f32 (seen in compiler output); no builtin is declared,
// so emit it directly. dst.lo = bf16(x), dst.hi = bf16(y).
__device__ __forceinline__ unsigned pkbf(float x, float y) {
  unsigned r;
  asm("v_cvt_pk_bf16_f32 %0, %1, %2" : "=v"(r) : "v"(x), "v"(y));
  return r;
}
__device__ __forceinline__ unsigned short sbf(float x) {
  return __builtin_bit_cast(unsigned short, (__bf16)x);
}

union FragBF { v16bf v; unsigned u[8]; };

// ---------------------------------------------------------------------------
// Gating: logits = x @ Wg + bg ; softmax over 10 ; mask routed cols to top-2.
// One wave per token. Writes dense gate matrix G[B][10].
// ---------------------------------------------------------------------------
__global__ __launch_bounds__(256) void gate_kernel(
    const float* __restrict__ x, const float* __restrict__ Wg,
    const float* __restrict__ bg, float* __restrict__ G) {
  const int lane = threadIdx.x & 31;
  const int b = blockIdx.x * 8 + (threadIdx.x >> 5);
  if (b >= B_TOK) return;
  float a[NEXP];
#pragma unroll
  for (int j = 0; j < NEXP; ++j) a[j] = 0.f;
  for (int d = lane; d < D_IN; d += 32) {
    float xv = x[(size_t)b * D_IN + d];
    const float* wr = Wg + (size_t)d * NEXP;
#pragma unroll
    for (int j = 0; j < NEXP; ++j) a[j] += xv * wr[j];
  }
#pragma unroll
  for (int j = 0; j < NEXP; ++j) {
#pragma unroll
    for (int off = 16; off > 0; off >>= 1)
      a[j] += __shfl_xor(a[j], off, 32);
  }
  if (lane == 0) {
    float mx = -3.4e38f;
#pragma unroll
    for (int j = 0; j < NEXP; ++j) { a[j] += bg[j]; mx = fmaxf(mx, a[j]); }
    float s = 0.f;
#pragma unroll
    for (int j = 0; j < NEXP; ++j) { a[j] = expf(a[j] - mx); s += a[j]; }
    float inv = 1.f / s;
#pragma unroll
    for (int j = 0; j < NEXP; ++j) a[j] *= inv;
    // top-2 among routed columns [2,10); ties pick lower index (matches top_k)
    int i1 = 2;
#pragma unroll
    for (int j = 3; j < NEXP; ++j) if (a[j] > a[i1]) i1 = j;
    int i2 = (i1 == 2) ? 3 : 2;
#pragma unroll
    for (int j = 2; j < NEXP; ++j) if (j != i1 && a[j] > a[i2]) i2 = j;
    float* gr = G + (size_t)b * NEXP;
#pragma unroll
    for (int j = 0; j < NEXP; ++j)
      gr[j] = (j < 2) ? a[j] : ((j == i1 || j == i2) ? a[j] : 0.f);
  }
}

// ---------------------------------------------------------------------------
// Tiled bf16 WMMA GEMM: C[M,N] = A[M,K] * Bw[K,N] (+bias, +epilogue)
//   A_F32: A is fp32 (converted to bf16 at staging) else raw bf16.
//   EPI=0: relu, store bf16 into Out (h buffer, ld = N)
//   EPI=1: Out(fp32, ld = N) += G[row][gcol] * (acc + bias)
// 256 threads (8 waves), block tile 128x128, K-step 32, software-pipelined
// global->reg prefetch so VMEM latency overlaps the WMMA stream.
// ---------------------------------------------------------------------------
template <bool A_F32, int EPI>
__global__ __launch_bounds__(256) void gemm_wmma(
    const void* __restrict__ A, const float* __restrict__ Bw,
    const float* __restrict__ bias, const float* __restrict__ G, int gcol,
    void* __restrict__ Out, int M, int K, int N) {
  __shared__ unsigned As[128 * 16];  // [row][kpair] : 128x32 bf16, dword-packed
  __shared__ unsigned Bs[128 * 16];  // [n][kpair]   : K-transposed, dword-packed

  const int tid  = threadIdx.x;
  const int lane = tid & 31;
  const int wave = tid >> 5;
  const int wm   = wave >> 2;   // 0..1 : 64-row slice
  const int wn   = wave & 3;    // 0..3 : 32-col slice
  const int half = lane >> 4;   // hi/lo lane group
  const int mr   = lane & 15;
  const int n0   = blockIdx.x * 128;
  const int m0   = blockIdx.y * 128;

  const v8f vzero = {0.f, 0.f, 0.f, 0.f, 0.f, 0.f, 0.f, 0.f};
  v8f acc[4][2];
#pragma unroll
  for (int mi = 0; mi < 4; ++mi)
#pragma unroll
    for (int ni = 0; ni < 2; ++ni) acc[mi][ni] = vzero;

  // prefetch registers (one K-tile ahead)
  float2   paf[8];   // A fp32 path
  unsigned pau[8];   // A bf16 path
  float    pbx[8], pby[8];

  auto prefetch = [&](int kk) {
#pragma unroll
    for (int i = 0; i < 8; ++i) {
      int s = tid + i * 256;          // 2048 A dword slots
      int row = s >> 4;
      int kp  = s & 15;
      if (A_F32)
        paf[i] = *(const float2*)((const float*)A +
                  (size_t)(m0 + row) * K + kk + kp * 2);
      else
        pau[i] = *(const unsigned*)((const unsigned short*)A +
                  (size_t)(m0 + row) * K + kk + kp * 2);
    }
#pragma unroll
    for (int i = 0; i < 8; ++i) {
      int s  = tid + i * 256;         // 2048 B dword slots
      int kp = s >> 7;                // 0..15 (covers K pair 2kp,2kp+1)
      int n  = s & 127;
      pbx[i] = Bw[(size_t)(kk + 2 * kp)     * N + n0 + n];
      pby[i] = Bw[(size_t)(kk + 2 * kp + 1) * N + n0 + n];
    }
  };
  auto stage = [&]() {
#pragma unroll
    for (int i = 0; i < 8; ++i) {
      int s = tid + i * 256;
      int row = s >> 4;
      int kp  = s & 15;
      As[row * 16 + kp] = A_F32 ? pkbf(paf[i].x, paf[i].y) : pau[i];
    }
#pragma unroll
    for (int i = 0; i < 8; ++i) {
      int s  = tid + i * 256;
      int kp = s >> 7;
      int n  = s & 127;
      Bs[n * 16 + kp] = pkbf(pbx[i], pby[i]);   // fragment-ready layout
    }
  };

  prefetch(0);
  for (int kk = 0; kk < K; kk += 32) {
    stage();
    __syncthreads();
    if (kk + 32 < K) prefetch(kk + 32);  // VMEM overlaps the WMMAs below

    // ---- fragments (layouts per CDNA5 ISA 7.12.2) ----
    FragBF af[4];
#pragma unroll
    for (int mi = 0; mi < 4; ++mi) {
      int row = wm * 64 + mi * 16 + mr;
#pragma unroll
      for (int j = 0; j < 8; ++j) {
        // VGPR j<4: K=2j(+8*half) ; j>=4: K=16+2(j-4)(+8*half)
        int kp = ((j < 4) ? j : (j + 4)) + 4 * half;
        af[mi].u[j] = As[row * 16 + kp];
      }
    }
    FragBF bfr[2];
#pragma unroll
    for (int ni = 0; ni < 2; ++ni) {
      int n = wn * 32 + ni * 16 + mr;
#pragma unroll
      for (int v = 0; v < 8; ++v)   // VGPR v<4: K=2v(+8*half) ; v>=4: K=16+2(v-4)(+8*half)
        bfr[ni].u[v] = Bs[n * 16 + ((v < 4) ? v : (v + 4)) + 4 * half];
    }

#pragma unroll
    for (int mi = 0; mi < 4; ++mi)
#pragma unroll
      for (int ni = 0; ni < 2; ++ni)
        acc[mi][ni] = __builtin_amdgcn_wmma_f32_16x16x32_bf16(
            false, af[mi].v, false, bfr[ni].v, (short)0, acc[mi][ni],
            false, false);
    __syncthreads();
  }

  // ---- epilogue ----
#pragma unroll
  for (int mi = 0; mi < 4; ++mi) {
#pragma unroll
    for (int ni = 0; ni < 2; ++ni) {
      int row0 = m0 + wm * 64 + mi * 16 + 8 * half;  // C/D: M = r + 8*half
      int col  = n0 + wn * 32 + ni * 16 + mr;        // N = lane&15
      float bv = bias[col];
#pragma unroll
      for (int r = 0; r < 8; ++r) {
        int row = row0 + r;
        float v = acc[mi][ni][r] + bv;
        if (EPI == 0) {
          v = fmaxf(v, 0.f);
          ((unsigned short*)Out)[(size_t)row * N + col] = sbf(v);
        } else {
          float g = G[(size_t)row * NEXP + gcol];
          float* op = (float*)Out + (size_t)row * N + col;
          *op += g * v;
        }
      }
    }
  }
}

// ---------------------------------------------------------------------------
extern "C" void kernel_launch(void* const* d_in, const int* in_sizes, int n_in,
                              void* d_out, int out_size, void* d_ws, size_t ws_size,
                              hipStream_t stream) {
  (void)in_sizes; (void)n_in; (void)out_size; (void)ws_size;
  const float* x   = (const float*)d_in[0];
  const float* W1  = (const float*)d_in[1];
  const float* b1  = (const float*)d_in[2];
  const float* W2  = (const float*)d_in[3];
  const float* b2  = (const float*)d_in[4];
  const float* Ws1 = (const float*)d_in[5];
  const float* bs1 = (const float*)d_in[6];
  const float* Ws2 = (const float*)d_in[7];
  const float* bs2 = (const float*)d_in[8];
  const float* Wg  = (const float*)d_in[9];
  const float* bg  = (const float*)d_in[10];
  float* out = (float*)d_out;

  // workspace: G[B][10] fp32 (327680 B, 256-aligned) then h[B][H] bf16 (64 MB)
  float* G = (float*)d_ws;
  unsigned short* h =
      (unsigned short*)((char*)d_ws + (size_t)B_TOK * NEXP * sizeof(float));

  hipMemsetAsync(out, 0, (size_t)B_TOK * O_DIM * sizeof(float), stream);
  gate_kernel<<<B_TOK / 8, 256, 0, stream>>>(x, Wg, bg, G);

  for (int e = 0; e < NEXP; ++e) {
    const float *w1, *bb1, *w2, *bb2;
    if (e < 2) {  // shared experts occupy gate columns 0..1
      w1  = Ws1 + (size_t)e * D_IN * H_DIM;
      bb1 = bs1 + (size_t)e * H_DIM;
      w2  = Ws2 + (size_t)e * H_DIM * O_DIM;
      bb2 = bs2 + (size_t)e * O_DIM;
    } else {      // routed experts occupy gate columns 2..9
      int i = e - 2;
      w1  = W1 + (size_t)i * D_IN * H_DIM;
      bb1 = b1 + (size_t)i * H_DIM;
      w2  = W2 + (size_t)i * H_DIM * O_DIM;
      bb2 = b2 + (size_t)i * O_DIM;
    }
    // h = relu(x @ w1 + bb1)   [8192 x 4096]
    gemm_wmma<true, 0><<<dim3(H_DIM / 128, B_TOK / 128), 256, 0, stream>>>(
        x, w1, bb1, nullptr, 0, h, B_TOK, D_IN, H_DIM);
    // out += G[:,e] * (h @ w2 + bb2)   [8192 x 1024]
    gemm_wmma<false, 1><<<dim3(O_DIM / 128, B_TOK / 128), 256, 0, stream>>>(
        h, w2, bb2, G, e, out, B_TOK, H_DIM, O_DIM);
  }
}
